// TopkGCNBlock_20744692039843
// MI455X (gfx1250) — matrix operand order repaired and checked by reading, not compiled, not run on hardware
//
#include <hip/hip_runtime.h>
#include <math.h>

// ---------------- problem constants (from reference) ----------------
#define GG     64
#define NN     1024
#define FF     128
#define ETOT   1048576          // G * 16384
#define NT     65536            // G * N
#define KK     820              // ceil(0.8 * N)
#define BN_EPS 1e-5f

typedef __bf16 bf16_t;
typedef __attribute__((ext_vector_type(4)))  __bf16 v4bf;
typedef __attribute__((ext_vector_type(8)))  __bf16 v8bf;
typedef __attribute__((ext_vector_type(16))) __bf16 v16bf;
typedef __attribute__((ext_vector_type(8)))  float  v8f;

// =====================================================================
// 0) init: deg = 1 (self loop weight), mapping = -1, stats = 0
// =====================================================================
__global__ void init_kernel(float* __restrict__ deg, int* __restrict__ mapping,
                            float* __restrict__ stats) {
  int i = blockIdx.x * blockDim.x + threadIdx.x;
  if (i < NT) { deg[i] = 1.0f; mapping[i] = -1; }
  if (i < 1024) stats[i] = 0.0f;
}

// =====================================================================
// 1) weighted in-degree scatter: deg[col] += w   (atomic f32)
// =====================================================================
__global__ void deg_scatter(const int* __restrict__ col,
                            const float* __restrict__ w,
                            float* __restrict__ deg) {
  int e = blockIdx.x * blockDim.x + threadIdx.x;
  if (e < ETOT) atomicAdd(&deg[col[e]], w[e]);
}

// =====================================================================
// 2) dis = rsqrt(deg)  (in place)
// =====================================================================
__global__ void dis_kernel(float* __restrict__ deg) {
  int i = blockIdx.x * blockDim.x + threadIdx.x;
  if (i < NT) deg[i] = rsqrtf(deg[i]);
}

// =====================================================================
// 3) xw = x @ W via V_WMMA_F32_16X16X32_BF16.
//    Block: 256 threads (8 wave32), 128-row M tile, full N=128.
//    LDS: bf16 tiles, row stride 136 elements (272 B = 17*16 B, so every
//    8-element fragment run is 16 B aligned => ds_load_b128), and
//    r*LDA dword address = r*68 -> r*4 mod 64 distinct banks for the
//    A-fragment column access pattern.
//    W is stored TRANSPOSED in LDS (Bt[col][k]) so each B fragment is
//    two contiguous 8-element runs -> 2x ds_load_b128 instead of 16x
//    scalar ds_load_u16.
// =====================================================================
#define LDA 136
__global__ __launch_bounds__(256) void gemm_xw_wmma(
    const float* __restrict__ x, const float* __restrict__ Wm,
    float* __restrict__ xw) {
  __shared__ __align__(16) bf16_t As[128 * LDA];  // x tile, row-major [m][k]
  __shared__ __align__(16) bf16_t Bt[128 * LDA];  // W transposed   [n][k]
  const int tid = threadIdx.x;
  const int m0  = blockIdx.x * 128;

  __builtin_prefetch(Wm, 0, 1);  // global_prefetch_b8: W is hot in L2 across all blocks

  // cooperative stage + fp32->bf16 convert (float4 granularity)
  for (int i = tid; i < 4096; i += 256) {          // 4096 float4 = 128x128 floats
    const int r = i >> 5;
    const int c = (i & 31) << 2;
    const float4 av = *(const float4*)(x  + (size_t)(m0 + r) * FF + c);
    const float4 bv = *(const float4*)(Wm + (size_t)r * FF + c);
    // A: 4 contiguous bf16 -> one 8-byte LDS store
    *(v4bf*)&As[r * LDA + c] =
        (v4bf){(bf16_t)av.x, (bf16_t)av.y, (bf16_t)av.z, (bf16_t)av.w};
    // B transposed: scatter 4 columns (one-time 64 KB transpose per block)
    Bt[(c + 0) * LDA + r] = (bf16_t)bv.x;
    Bt[(c + 1) * LDA + r] = (bf16_t)bv.y;
    Bt[(c + 2) * LDA + r] = (bf16_t)bv.z;
    Bt[(c + 3) * LDA + r] = (bf16_t)bv.w;
  }
  __syncthreads();

  const int wave = tid >> 5;
  const int lane = tid & 31;
  const int half = lane >> 4;       // 0: lanes 0-15, 1: lanes 16-31
  const int l16  = lane & 15;
  const int rowA = wave * 16 + l16; // this lane's A row within the 128-row tile

  v8f acc[8];
#pragma unroll
  for (int n = 0; n < 8; ++n)
#pragma unroll
    for (int i = 0; i < 8; ++i) acc[n][i] = 0.0f;

#pragma unroll
  for (int kk = 0; kk < 128; kk += 32) {
    // A fragment, 16-bit A 16x32 layout:
    //   e<8 : K = kk + half*8 + e ; e>=8 : K = kk + 16 + half*8 + (e-8)
    const v8bf alo = *(const v8bf*)&As[rowA * LDA + kk + half * 8];
    const v8bf ahi = *(const v8bf*)&As[rowA * LDA + kk + 16 + half * 8];
    const v16bf a = __builtin_shufflevector(alo, ahi,
        0, 1, 2, 3, 4, 5, 6, 7, 8, 9, 10, 11, 12, 13, 14, 15);
#pragma unroll
    for (int n = 0; n < 8; ++n) {
      // B fragment, same K order as A: e<8 : K = kk + half*8 + e ; e>=8 : K = kk + 16 + half*8 + (e-8)
      const bf16_t* bp = &Bt[(n * 16 + l16) * LDA + kk + half * 8];
      const v8bf blo = *(const v8bf*)(bp);
      const v8bf bhi = *(const v8bf*)(bp + 16);
      const v16bf b = __builtin_shufflevector(blo, bhi,
          0, 1, 2, 3, 4, 5, 6, 7, 8, 9, 10, 11, 12, 13, 14, 15);
      acc[n] = __builtin_amdgcn_wmma_f32_16x16x32_bf16(
          false, a, false, b, (short)0, acc[n], false, false);
    }
  }

  // C/D layout: VGPR i -> M = i + half*8, col = n*16 + l16
#pragma unroll
  for (int n = 0; n < 8; ++n)
#pragma unroll
    for (int i = 0; i < 8; ++i) {
      const int m = wave * 16 + half * 8 + i;
      xw[(size_t)(m0 + m) * FF + n * 16 + l16] = acc[n][i];
    }
}

// =====================================================================
// 4) agg = b + (1/deg) * xw     (self-loop term; dis holds rsqrt(deg))
// =====================================================================
__global__ void agg_init(const float* __restrict__ xw, const float* __restrict__ dis,
                         const float* __restrict__ bias, float* __restrict__ agg) {
  const int i = blockIdx.x, f = threadIdx.x;
  const float d = dis[i];
  agg[(size_t)i * FF + f] = bias[f] + xw[(size_t)i * FF + f] * d * d;
}

// =====================================================================
// 5) edge SpMM: agg[col] += dis[row]*w*dis[col] * xw[row]
//    one wave per edge, lane owns 4 features (float4 gather, 4 atomics)
// =====================================================================
__global__ __launch_bounds__(128) void edge_spmm(
    const int* __restrict__ row, const int* __restrict__ col,
    const float* __restrict__ w, const float* __restrict__ dis,
    const float* __restrict__ xw, float* __restrict__ agg) {
  const int e    = blockIdx.x * 4 + (threadIdx.x >> 5);
  const int lane = threadIdx.x & 31;
  const int r = row[e], c = col[e];
  const float coef = dis[r] * w[e] * dis[c];
  const float4 xv = *(const float4*)(xw + (size_t)r * FF + lane * 4);
  float* dst = agg + (size_t)c * FF + lane * 4;
  atomicAdd(dst + 0, coef * xv.x);
  atomicAdd(dst + 1, coef * xv.y);
  atomicAdd(dst + 2, coef * xv.z);
  atomicAdd(dst + 3, coef * xv.w);
}

// =====================================================================
// 6) BN stats: per-feature sum / sumsq (coalesced, strided rows)
// =====================================================================
__global__ void bn_stats(const float* __restrict__ agg, float* __restrict__ stats) {
  const int f = threadIdx.x;
  float s = 0.f, q = 0.f;
  for (int r = blockIdx.x; r < NT; r += 256) {
    const float v = agg[(size_t)r * FF + f];
    s += v; q += v * v;
  }
  atomicAdd(&stats[f], s);
  atomicAdd(&stats[128 + f], q);
}

// =====================================================================
// 7) BN finalize (mu, rstd) + 1/||p||  (single block)
// =====================================================================
__global__ void bn_finalize(float* __restrict__ stats, const float* __restrict__ p) {
  __shared__ float red[128];
  const int f = threadIdx.x;
  const float mu  = stats[f] * (1.0f / (float)NT);
  const float var = stats[128 + f] * (1.0f / (float)NT) - mu * mu;
  stats[256 + f] = mu;
  stats[384 + f] = rsqrtf(var + BN_EPS);
  red[f] = p[f] * p[f];
  __syncthreads();
  for (int s = 64; s > 0; s >>= 1) { if (f < s) red[f] += red[f + s]; __syncthreads(); }
  if (f == 0) stats[512] = rsqrtf(red[0]);   // 1/||p||
}

// =====================================================================
// 8) BN apply + ReLU (h written in place over agg) + score = h.p/||p||
// =====================================================================
__global__ void bn_apply(float* __restrict__ agg, const float* __restrict__ stats,
                         const float* __restrict__ gamma, const float* __restrict__ beta,
                         const float* __restrict__ p, float* __restrict__ score) {
  __shared__ float red[128];
  const int i = blockIdx.x, f = threadIdx.x;
  float h = (agg[(size_t)i * FF + f] - stats[256 + f]) * stats[384 + f] * gamma[f] + beta[f];
  h = fmaxf(h, 0.0f);
  agg[(size_t)i * FF + f] = h;
  red[f] = h * p[f];
  __syncthreads();
  for (int s = 64; s > 0; s >>= 1) { if (f < s) red[f] += red[f + s]; __syncthreads(); }
  if (f == 0) score[i] = red[0] * stats[512];
}

// =====================================================================
// 9) per-graph descending top-K via full bitonic sort of N=1024 in LDS
//    comparator: (val desc, idx asc) -> matches jax.lax.top_k ordering
// =====================================================================
__global__ __launch_bounds__(1024) void topk_kernel(
    const float* __restrict__ score, float* __restrict__ vals,
    int* __restrict__ idxo, int* __restrict__ mapping) {
  __shared__ float sv[NN];
  __shared__ int   si[NN];
  const int g = blockIdx.x, t = threadIdx.x;
  sv[t] = score[g * NN + t];
  si[t] = t;
  __syncthreads();
  for (int k = 2; k <= NN; k <<= 1) {
    for (int j = k >> 1; j > 0; j >>= 1) {
      const int ixj = t ^ j;
      if (ixj > t) {
        const bool dir = ((t & k) == 0);
        const float v0 = sv[t], v1 = sv[ixj];
        const int   i0 = si[t], i1 = si[ixj];
        const bool lt = (v1 > v0) || (v1 == v0 && i1 < i0); // [ixj] comes-before [t]
        if (lt == dir) { sv[t] = v1; sv[ixj] = v0; si[t] = i1; si[ixj] = i0; }
      }
      __syncthreads();
    }
  }
  if (t < KK) {
    vals[g * KK + t] = sv[t];
    idxo[g * KK + t] = si[t];
    mapping[g * NN + si[t]] = g * KK + t;
  }
}

// =====================================================================
// 10) h_pool[gk] = h[g*N + idx[gk]] * tanh(vals[gk])   -> d_out section 0
// =====================================================================
__global__ void pool_gather(const float* __restrict__ h, const float* __restrict__ vals,
                            const int* __restrict__ idxo, float* __restrict__ out_hpool) {
  const int gk = blockIdx.x, f = threadIdx.x;
  const int g = gk / KK;
  const int src = g * NN + idxo[gk];
  const float gate = tanhf(vals[gk]);
  out_hpool[(size_t)gk * FF + f] = h[(size_t)src * FF + f] * gate;
}

// =====================================================================
// 11) per-graph mean + max over K pooled rows -> flat [G, 2F]
// =====================================================================
__global__ void graph_pool(const float* __restrict__ hpool, float* __restrict__ flat) {
  const int g = blockIdx.x, f = threadIdx.x;
  float s = 0.f, m = -INFINITY;
  for (int k = 0; k < KK; ++k) {
    const float v = hpool[((size_t)g * KK + k) * FF + f];
    s += v;
    m = fmaxf(m, v);
  }
  flat[g * 2 * FF + f]      = s * (1.0f / (float)KK);
  flat[g * 2 * FF + FF + f] = m;
}

// =====================================================================
// 12) edge remap (+ batch_new); int outputs stored as float values
// =====================================================================
__global__ void edge_remap(const int* __restrict__ row, const int* __restrict__ col,
                           const float* __restrict__ w, const int* __restrict__ mapping,
                           float* __restrict__ ei_new, float* __restrict__ ew_new,
                           float* __restrict__ batch_new) {
  const int e = blockIdx.x * blockDim.x + threadIdx.x;
  if (e < ETOT) {
    const int nr = mapping[row[e]];
    const int nc = mapping[col[e]];
    const bool valid = (nr >= 0) && (nc >= 0);
    ei_new[e]        = (float)(valid ? nr : 0);
    ei_new[ETOT + e] = (float)(valid ? nc : 0);
    ew_new[e]        = valid ? w[e] : 0.0f;
  }
  if (e < GG * KK) batch_new[e] = (float)(e / KK);
}

// =====================================================================
// launcher
// =====================================================================
extern "C" void kernel_launch(void* const* d_in, const int* in_sizes, int n_in,
                              void* d_out, int out_size, void* d_ws, size_t ws_size,
                              hipStream_t stream) {
  (void)in_sizes; (void)n_in; (void)out_size; (void)ws_size;
  const float* x     = (const float*)d_in[0];
  const float* ew    = (const float*)d_in[1];
  const float* Wm    = (const float*)d_in[2];
  const float* bias  = (const float*)d_in[3];
  const float* gamma = (const float*)d_in[4];
  const float* beta  = (const float*)d_in[5];
  const float* p     = (const float*)d_in[6];
  const int*   eidx  = (const int*)d_in[7];
  const int*   erow  = eidx;
  const int*   ecol  = eidx + ETOT;

  // ---- workspace layout ----
  float* ws      = (float*)d_ws;
  float* xw      = ws;                                 // NT*F
  float* hbuf    = xw + (size_t)NT * FF;               // NT*F (agg -> h in place)
  float* deg     = hbuf + (size_t)NT * FF;             // NT (-> dis)
  float* score   = deg + NT;                           // NT
  float* stats   = score + NT;                         // 1024
  float* vals    = stats + 1024;                       // G*K
  int*   idxbuf  = (int*)(vals + (size_t)GG * KK);     // G*K
  int*   mapping = idxbuf + (size_t)GG * KK;           // NT

  // ---- output layout (flat concat in reference return order) ----
  float* out      = (float*)d_out;
  float* o_hpool  = out;                                   // G*K*F
  float* o_flat   = o_hpool + (size_t)GG * KK * FF;        // G*2F
  float* o_ei     = o_flat + (size_t)GG * 2 * FF;          // 2*E
  float* o_ew     = o_ei + (size_t)2 * ETOT;               // E
  float* o_batch  = o_ew + (size_t)ETOT;                   // G*K

  init_kernel<<<NT / 256, 256, 0, stream>>>(deg, mapping, stats);
  deg_scatter<<<ETOT / 256, 256, 0, stream>>>(ecol, ew, deg);
  dis_kernel<<<NT / 256, 256, 0, stream>>>(deg);
  gemm_xw_wmma<<<NT / 128, 256, 0, stream>>>(x, Wm, xw);
  agg_init<<<NT, FF, 0, stream>>>(xw, deg, bias, hbuf);
  edge_spmm<<<ETOT / 4, 128, 0, stream>>>(erow, ecol, ew, deg, xw, hbuf);
  bn_stats<<<256, FF, 0, stream>>>(hbuf, stats);
  bn_finalize<<<1, FF, 0, stream>>>(stats, p);
  bn_apply<<<NT, FF, 0, stream>>>(hbuf, stats, gamma, beta, p, score);
  topk_kernel<<<GG, NN, 0, stream>>>(score, vals, idxbuf, mapping);
  pool_gather<<<GG * KK, FF, 0, stream>>>(hbuf, vals, idxbuf, o_hpool);
  graph_pool<<<GG, FF, 0, stream>>>(o_hpool, o_flat);
  edge_remap<<<ETOT / 256, 256, 0, stream>>>(erow, ecol, ew, mapping,
                                             o_ei, o_ew, o_batch);
}
